// DeformableBlock_33346126086528
// MI455X (gfx1250) — matrix operand order repaired and checked by reading, not compiled, not run on hardware
//
#include <hip/hip_runtime.h>
#include <hip/hip_bf16.h>

typedef __attribute__((ext_vector_type(16))) __bf16 v16bf;
typedef __attribute__((ext_vector_type(8)))  __bf16 v8bf;
typedef __attribute__((ext_vector_type(8)))  float  v8f;

__device__ __forceinline__ int iclamp(int v, int lo, int hi) {
    return v < lo ? lo : (v > hi ? hi : v);
}

// ---------------------------------------------------------------------------
// Kernel 1: offset = conv2d(x, w_off, b_off, pad=1)   (128 -> 18 channels)
// ---------------------------------------------------------------------------
__global__ __launch_bounds__(256)
void offset_conv_kernel(const float* __restrict__ x,
                        const float* __restrict__ w_off,
                        const float* __restrict__ b_off,
                        float* __restrict__ offset) {
    const int bh = blockIdx.x;          // 0..511
    const int b  = bh >> 6;
    const int h  = bh & 63;
    const int t  = threadIdx.x;
    const int w  = t & 63;
    const int oc0 = t >> 6;             // 0..3
    const float* xb = x + (size_t)b * 128 * 4096;

    for (int oc = oc0; oc < 18; oc += 4) {
        float acc = b_off[oc];
        const float* wf = w_off + (size_t)oc * 1152;
        for (int c = 0; c < 128; ++c) {
            const float* xp = xb + (size_t)c * 4096;
            const float* wc = wf + c * 9;
            #pragma unroll
            for (int ky = 0; ky < 3; ++ky) {
                const int y = h + ky - 1;
                if (y < 0 || y >= 64) continue;
                const float* xr = xp + y * 64;
                #pragma unroll
                for (int kx = 0; kx < 3; ++kx) {
                    const int xx = w + kx - 1;
                    if (xx < 0 || xx >= 64) continue;
                    acc = fmaf(xr[xx], wc[ky * 3 + kx], acc);
                }
            }
        }
        offset[(((size_t)b * 18 + oc) * 64 + h) * 64 + w] = acc;
    }
}

// ---------------------------------------------------------------------------
// Kernel 2: wkT[k'][n] = bf16(w_def[n][c*9+kk]) with PERMUTED K: k' = kk*128+c.
// The permutation makes every 32-wide K-chunk of the GEMM have a constant
// kernel-point kk (geometry loaded once per chunk, gather base advances by one
// channel plane per element). GEMM result is invariant to K ordering.
// ---------------------------------------------------------------------------
__global__ void wdefT_bf16_kernel(const float* __restrict__ w_def,
                                  __bf16* __restrict__ wkT) {
    int i = blockIdx.x * blockDim.x + threadIdx.x;   // flat over [1152][256]
    if (i < 1152 * 256) {
        const int kp = i >> 8;          // permuted k' = kk*128 + c
        const int n  = i & 255;
        const int kk = kp >> 7;
        const int c  = kp & 127;
        wkT[i] = (__bf16)w_def[(size_t)n * 1152 + c * 9 + kk];
    }
}

// ---------------------------------------------------------------------------
// Kernel 3: fused deformable gather + GEMM (bf16 WMMA, f32 acc) + bias + ReLU
// Tile: M=64 (one image row), N=256 (all Cout -> gather done once per element).
// 8 waves; wave w owns N-range [w*32, w*32+32): 8 WMMA accs per wave.
// K' = 1152 in 36 chunks of 32; chunk it: kk = it>>2, c0 = (it&3)*32.
// Double-buffered LDS tiles: one barrier per K-step, gather overlaps WMMA.
// ---------------------------------------------------------------------------
__global__ __launch_bounds__(256)
void deform_wmma_kernel(const float* __restrict__ x,
                        const float* __restrict__ offset,
                        const __bf16* __restrict__ wkT,
                        const float* __restrict__ b_def,
                        float* __restrict__ out) {
    __shared__ int    sOff[576][4];        // bilinear corner offsets, clamped
    __shared__ float  sW[576][4];          // bilinear weights, pre-masked for OOB
    __shared__ __bf16 sA[2][64][40];       // A tiles: 64 m x 32 k (80B rows)
    __shared__ __bf16 sB[2][32][264];      // B tiles: 32 k x 256 n (528B rows)
    __shared__ float  sOut[32][66];        // epilogue transpose buffer

    const int t  = threadIdx.x;
    const int m0 = blockIdx.x * 64;        // one image row: b,h uniform, w=ml
    const int b  = m0 >> 12;
    const int h  = (m0 >> 6) & 63;
    const float* xb = x + (size_t)b * 128 * 4096;

    // ---- Stage 0: bilinear sample geometry (reused across all 128 channels)
    for (int idx = t; idx < 576; idx += 256) {
        const int ml = idx / 9;
        const int kk = idx - ml * 9;
        const int w  = ml;
        const float dy = offset[(((size_t)b * 18 + 2 * kk    ) * 64 + h) * 64 + w];
        const float dx = offset[(((size_t)b * 18 + 2 * kk + 1) * 64 + h) * 64 + w];
        const float py = (float)(h + kk / 3 - 1) + dy;
        const float px = (float)(w + kk % 3 - 1) + dx;
        const float y0f = floorf(py), x0f = floorf(px);
        const float ly = py - y0f, lx = px - x0f;
        const int y0 = (int)y0f, x0 = (int)x0f;
        const float wbl[4] = { (1.f - ly) * (1.f - lx), (1.f - ly) * lx,
                               ly * (1.f - lx),         ly * lx };
        const int yy[4] = { y0, y0, y0 + 1, y0 + 1 };
        const int xx[4] = { x0, x0 + 1, x0, x0 + 1 };
        #pragma unroll
        for (int i = 0; i < 4; ++i) {
            const bool v = (yy[i] >= 0) & (yy[i] < 64) & (xx[i] >= 0) & (xx[i] < 64);
            sOff[idx][i] = iclamp(yy[i], 0, 63) * 64 + iclamp(xx[i], 0, 63);
            sW[idx][i]   = v ? wbl[i] : 0.f;
        }
    }
    __syncthreads();

    const int wave   = t >> 5;             // 0..7 -> N-range wave*32
    const int lane   = t & 31;
    const int half   = lane >> 4;
    const int lane16 = lane & 15;

    v8f acc[4][2] = {};                    // [M-subtile][N-subtile]

    // A staging: thread -> (m row, 8 consecutive channels at fixed kk)
    const int s_ml = t >> 2;
    const int s_k0 = (t & 3) * 8;
    // B staging: thread -> (k row, 32 consecutive n) = one 64B chunk
    const int br  = t >> 3;                // 0..31
    const int bc0 = (t & 7) * 32;          // 0..224

    // Stage K-chunk `it` into LDS buffer `p`. Chunk has constant kk = it>>2,
    // channels c = (it&3)*32 + s_k0 + j: gather base advances one plane per j.
    auto stage = [&](int it, int p) {
        const int kk = it >> 2;
        const int bi = s_ml * 9 + kk;
        const int4   o4 = *(const int4*)sOff[bi];
        const float4 w4 = *(const float4*)sW[bi];
        const float* xp = xb + (size_t)(((it & 3) << 5) + s_k0) * 4096;
        v8bf av;
        #pragma unroll
        for (int j = 0; j < 8; ++j) {
            const float v = w4.x * xp[o4.x] + w4.y * xp[o4.y]
                          + w4.z * xp[o4.z] + w4.w * xp[o4.w];
            av[j] = (__bf16)v;
            xp += 4096;
        }
        *(v8bf*)&sA[p][s_ml][s_k0] = av;

        const uint4* src = (const uint4*)(wkT + (size_t)(it * 32 + br) * 256 + bc0);
        uint4* dst = (uint4*)&sB[p][br][bc0];
        #pragma unroll
        for (int j = 0; j < 4; ++j) dst[j] = src[j];
        if (it + 1 < 36)
            __builtin_prefetch(wkT + (size_t)((it + 1) * 32 + br) * 256 + bc0, 0, 1);
    };

    stage(0, 0);

    for (int it = 0; it < 36; ++it) {
        const int p = it & 1;
        __syncthreads();                   // buf p ready; buf p^1 free
        if (it + 1 < 36) stage(it + 1, p ^ 1);   // overlap next gather with WMMA

        // B fragments (N = wave*32 + j*16 + lane16, K = half*8 + e / 16 + half*8 + e)
        v16bf bfrag[2];
        #pragma unroll
        for (int j = 0; j < 2; ++j) {
            const int ncol = wave * 32 + j * 16 + lane16;
            #pragma unroll
            for (int e = 0; e < 8; ++e) { bfrag[j][e] = sB[p][half * 8 + e][ncol]; bfrag[j][e + 8] = sB[p][16 + half * 8 + e][ncol]; }
        }
        // A fragments per M-subtile, 2 WMMAs each
        #pragma unroll
        for (int wm = 0; wm < 4; ++wm) {
            const __bf16* ar = &sA[p][wm * 16 + lane16][0];
            v8bf alo = *(const v8bf*)(ar + half * 8);        // K = half*8 + 0..7
            v8bf ahi = *(const v8bf*)(ar + 16 + half * 8);   // K = 16 + half*8 + 0..7
            v16bf afrag;
            #pragma unroll
            for (int e = 0; e < 8; ++e) { afrag[e] = alo[e]; afrag[e + 8] = ahi[e]; }
            acc[wm][0] = __builtin_amdgcn_wmma_f32_16x16x32_bf16(
                             false, afrag, false, bfrag[0], (short)0, acc[wm][0], false, false);
            acc[wm][1] = __builtin_amdgcn_wmma_f32_16x16x32_bf16(
                             false, afrag, false, bfrag[1], (short)0, acc[wm][1], false, false);
        }
    }

    // ---- epilogue: 8 wave-rounds through LDS transpose; bias+ReLU fused;
    //      coalesced b128 row stores (out[b][n][h][0..63] is contiguous).
    for (int g = 0; g < 8; ++g) {
        __syncthreads();
        if (wave == g) {
            #pragma unroll
            for (int wm = 0; wm < 4; ++wm)
                #pragma unroll
                for (int j = 0; j < 2; ++j)
                    #pragma unroll
                    for (int i = 0; i < 8; ++i)
                        sOut[j * 16 + lane16][wm * 16 + half * 8 + i] = acc[wm][j][i];
        }
        __syncthreads();
        #pragma unroll
        for (int r = 0; r < 2; ++r) {
            const int id = t * 2 + r;                // 0..511 = 32 n-rows x 16 float4
            const int nl = id >> 4;
            const int q  = id & 15;
            const int n  = g * 32 + nl;
            const float bias = b_def[n];
            float4 v;
            v.x = fmaxf(sOut[nl][q * 4 + 0] + bias, 0.f);
            v.y = fmaxf(sOut[nl][q * 4 + 1] + bias, 0.f);
            v.z = fmaxf(sOut[nl][q * 4 + 2] + bias, 0.f);
            v.w = fmaxf(sOut[nl][q * 4 + 3] + bias, 0.f);
            float* op = out + (((size_t)b * 256 + n) * 4096 + h * 64) + q * 4;
            *(float4*)op = v;
        }
    }
}

// ---------------------------------------------------------------------------
extern "C" void kernel_launch(void* const* d_in, const int* in_sizes, int n_in,
                              void* d_out, int out_size, void* d_ws, size_t ws_size,
                              hipStream_t stream) {
    const float* x     = (const float*)d_in[0];   // [8,128,64,64]
    const float* w_off = (const float*)d_in[1];   // [18,128,3,3]
    const float* b_off = (const float*)d_in[2];   // [18]
    const float* w_def = (const float*)d_in[3];   // [256,128,3,3]
    const float* b_def = (const float*)d_in[4];   // [256]
    float* out = (float*)d_out;                   // [8,256,64,64]

    float*  offs = (float*)d_ws;                                    // 589824 f32
    __bf16* wkT  = (__bf16*)((char*)d_ws + 589824 * sizeof(float)); // [1152][256] bf16 (permuted K)

    offset_conv_kernel<<<dim3(8 * 64), 256, 0, stream>>>(x, w_off, b_off, offs);
    wdefT_bf16_kernel<<<dim3((1152 * 256 + 255) / 256), 256, 0, stream>>>(w_def, wkT);
    deform_wmma_kernel<<<dim3(512), 256, 0, stream>>>(x, offs, wkT, b_def, out);
}
